// MultiheadAttention_28930899706447
// MI455X (gfx1250) — matrix operand order repaired and checked by reading, not compiled, not run on hardware
//
#include <hip/hip_runtime.h>
#include <hip/hip_bf16.h>

// MI455X / gfx1250, wave32. All matmuls via v_wmma_f32_16x16x32_bf16.
// T=2048 B=4 E=512 H=8 D=64. Ping-pong double-buffered fragment loads,
// loops kept rolled so the double buffer survives register allocation.

#define T_DIM 2048
#define B_DIM 4
#define E_DIM 512
#define H_DIM 8
#define D_DIM 64

typedef __bf16 bf16_t;
typedef __bf16  v16bf __attribute__((ext_vector_type(16)));
typedef __bf16  v8bf  __attribute__((ext_vector_type(8)));
typedef float   v8f   __attribute__((ext_vector_type(8)));

__device__ __forceinline__ bf16_t f2bf(float f) {
    unsigned u = __builtin_bit_cast(unsigned, f);
    unsigned r = u + 0x7FFFu + ((u >> 16) & 1u);   // round-to-nearest-even
    unsigned short h = (unsigned short)(r >> 16);
    return __builtin_bit_cast(bf16_t, h);
}

__device__ __forceinline__ v16bf join8(v8bf a, v8bf b) {
    v16bf r;
#pragma unroll
    for (int i = 0; i < 8; ++i) { r[i] = a[i]; r[i + 8] = b[i]; }
    return r;
}

__device__ __forceinline__ v8f wmma_bf16(v16bf a, v16bf b, v8f c) {
    // D(16x16,f32) = A(16x32,bf16) x B(32x16,bf16) + C
    return __builtin_amdgcn_wmma_f32_16x16x32_bf16(
        false, a, false, b, (short)0, c, false, false);
}

// ---------------------------------------------------------------- conversions
__global__ void cvt_f32_bf16(const float* __restrict__ src,
                             bf16_t* __restrict__ dst, int n) {
    int i = blockIdx.x * blockDim.x + threadIdx.x;
    if (i < n) dst[i] = f2bf(src[i]);
}

// ---------------------------------------------------------------- GEMM
// C(N=8192, 512) = X(8192,512) @ W(512,512)^T + bias, optional scale.
// mode 0: store bf16 head-major (b*H+h, t, d)         (Q, K)
// mode 1: store bf16 head-major transposed (bh, d, t)  (V^T)
// mode 2: store f32 row-major (n, e)                   (final out)
__global__ __launch_bounds__(256)
void gemm_wmma(const bf16_t* __restrict__ X, const bf16_t* __restrict__ W,
               const float* __restrict__ bias, float scale,
               bf16_t* __restrict__ dst_bf, float* __restrict__ dst_f32,
               int mode) {
    const int lane = threadIdx.x & 31;
    const int wave = threadIdx.x >> 5;
    const int hi   = lane >> 4;       // 0: lanes 0-15, 1: lanes 16-31
    const int ln   = lane & 15;
    const int row0 = blockIdx.x * 128 + wave * 16;
    const int c0   = blockIdx.y * 64;

    const bf16_t* Xrow = X + (size_t)(row0 + ln) * E_DIM;

    auto loadA = [&](int k0) -> v16bf {
        const bf16_t* pa = Xrow + k0 + hi * 8;
        return join8(*(const v8bf*)pa, *(const v8bf*)(pa + 16));
    };
    auto loadB = [&](int j, int k0) -> v16bf {
        const bf16_t* pb = W + (size_t)(c0 + j * 16 + ln) * E_DIM + k0 + hi * 8;
        return join8(*(const v8bf*)pb, *(const v8bf*)(pb + 16));
    };

    const v8f zero = {0.f, 0.f, 0.f, 0.f, 0.f, 0.f, 0.f, 0.f};
    v8f acc[4];
#pragma unroll
    for (int j = 0; j < 4; ++j) acc[j] = zero;

    // ping-pong double buffer: compute with one set, load into the other.
    // Loop must stay rolled so the two sets remain distinct loop-carried regs.
    v16bf af[2];
    v16bf bfr[2][4];
    af[0] = loadA(0);
#pragma unroll
    for (int j = 0; j < 4; ++j) bfr[0][j] = loadB(j, 0);

#pragma unroll 1
    for (int k0 = 0; k0 < E_DIM; k0 += 64) {
        // phase 0: load set1 @ k0+32, compute set0 @ k0
        af[1] = loadA(k0 + 32);
#pragma unroll
        for (int j = 0; j < 4; ++j) bfr[1][j] = loadB(j, k0 + 32);
#pragma unroll
        for (int j = 0; j < 4; ++j) acc[j] = wmma_bf16(af[0], bfr[0][j], acc[j]);
        // phase 1: load set0 @ (k0+64) mod E (wrap is harmless), compute set1
        const int k2 = (k0 + 64) & (E_DIM - 1);
        af[0] = loadA(k2);
#pragma unroll
        for (int j = 0; j < 4; ++j) bfr[0][j] = loadB(j, k2);
#pragma unroll
        for (int j = 0; j < 4; ++j) acc[j] = wmma_bf16(af[1], bfr[1][j], acc[j]);
    }

#pragma unroll
    for (int j = 0; j < 4; ++j) {
        const int col = c0 + j * 16 + ln;
        const float bv = bias[col];
#pragma unroll
        for (int r = 0; r < 8; ++r) {
            const int nrow = row0 + r + 8 * hi;     // C layout: m = r + 8*hi
            float v = (acc[j][r] + bv) * scale;
            if (mode == 2) {
                dst_f32[(size_t)nrow * E_DIM + col] = v;
            } else {
                const int t = nrow >> 2, b = nrow & 3;   // nrow = t*B + b
                const int h = col >> 6, d = col & 63;    // col  = h*D + d
                const bf16_t bvv = f2bf(v);
                if (mode == 0)
                    dst_bf[(((size_t)(b * H_DIM + h)) * T_DIM + t) * D_DIM + d] = bvv;
                else
                    dst_bf[(((size_t)(b * H_DIM + h)) * D_DIM + d) * T_DIM + t] = bvv;
            }
        }
    }
}

// ---------------------------------------------------------------- attention
// Per wave: one 16-row q tile of one head; flash-style over s in chunks of 32.
// K fragments + bias ping-pong one chunk ahead; V fragments are issued at the
// top of each chunk and consumed only after softmax.
__global__ __launch_bounds__(128)
void attn_wmma(const bf16_t* __restrict__ Q, const bf16_t* __restrict__ K,
               const bf16_t* __restrict__ VT, const float* __restrict__ bias,
               bf16_t* __restrict__ attn_out) {
    __shared__ __align__(16) bf16_t pbuf[4][16][40];  // per-wave P tile, padded

    const int lane = threadIdx.x & 31;
    const int wave = threadIdx.x >> 5;
    const int hi   = lane >> 4;
    const int ln   = lane & 15;
    const int bh   = blockIdx.y;                      // 0..31
    const int t0   = (blockIdx.x * 4 + wave) * 16;    // q-row tile base
    const int b    = bh >> 3, h = bh & 7;             // bh = b*H + h

    const bf16_t* Qh = Q  + (size_t)bh * T_DIM * D_DIM;
    const bf16_t* Kh = K  + (size_t)bh * T_DIM * D_DIM;
    const bf16_t* Vh = VT + (size_t)bh * D_DIM * T_DIM;
    const float*  Bh = bias + (size_t)bh * T_DIM * T_DIM;

    // Preload Q A-fragments (16x64 = two 16x32 fragments), kept in VGPRs.
    v16bf qf[2];
#pragma unroll
    for (int kk = 0; kk < 2; ++kk) {
        const bf16_t* p = Qh + (size_t)(t0 + ln) * D_DIM + kk * 32 + hi * 8;
        qf[kk] = join8(*(const v8bf*)p, *(const v8bf*)(p + 16));
    }

    v16bf kf[2][2][2];       // [buf][jtile][kstep]
    float bb[2][2][8];       // [buf][jtile][row]

    auto loadK = [&](int s0, int buf) {
#pragma unroll
        for (int j = 0; j < 2; ++j)
#pragma unroll
            for (int kk = 0; kk < 2; ++kk) {
                const bf16_t* pk = Kh + (size_t)(s0 + j * 16 + ln) * D_DIM +
                                   kk * 32 + hi * 8;
                kf[buf][j][kk] = join8(*(const v8bf*)pk, *(const v8bf*)(pk + 16));
            }
    };
    auto loadBias = [&](int s0, int buf) {
#pragma unroll
        for (int j = 0; j < 2; ++j)
#pragma unroll
            for (int r = 0; r < 8; ++r)
                bb[buf][j][r] =
                    Bh[(size_t)(t0 + r + 8 * hi) * T_DIM + s0 + j * 16 + ln];
    };

    const v8f zero = {0.f, 0.f, 0.f, 0.f, 0.f, 0.f, 0.f, 0.f};
    v8f o[4];
#pragma unroll
    for (int j = 0; j < 4; ++j) o[j] = zero;
    float m_run[8], l_run[8];
#pragma unroll
    for (int r = 0; r < 8; ++r) { m_run[r] = -3.0e38f; l_run[r] = 0.f; }

    loadK(0, 0);
    loadBias(0, 0);

    auto chunk = [&](int s0, int cur, int nxt) {
        // ---- V fragments for this chunk (consumed after softmax)
        v16bf vf[4];
#pragma unroll
        for (int j = 0; j < 4; ++j) {
            const bf16_t* pv = Vh + (size_t)(j * 16 + ln) * T_DIM + s0 + hi * 8;
            vf[j] = join8(*(const v8bf*)pv, *(const v8bf*)(pv + 16));
        }
        // ---- prefetch next chunk's K fragments + bias into the other buffer
        const int sn = s0 + 32;
        if (sn < T_DIM) {
            loadK(sn, nxt);
            loadBias(sn, nxt);
            __builtin_prefetch(&Bh[(size_t)(t0 + ln + 8 * hi) * T_DIM + sn + 32],
                               0, 0);
        }

        // ---- S(16x32) = Q(16x64) @ K^T(64x32): two 16x16 N-tiles
        v8f s[2];
        s[0] = zero; s[1] = zero;
#pragma unroll
        for (int j = 0; j < 2; ++j)
#pragma unroll
            for (int kk = 0; kk < 2; ++kk)
                s[j] = wmma_bf16(qf[kk], kf[cur][j][kk], s[j]);

        // ---- + bias (preloaded one chunk ahead)
#pragma unroll
        for (int j = 0; j < 2; ++j)
#pragma unroll
            for (int r = 0; r < 8; ++r)
                s[j][r] += bb[cur][j][r];

        // ---- online softmax (row spans 16 lanes of one VGPR slot)
        float alpha[8];
#pragma unroll
        for (int r = 0; r < 8; ++r) {
            float mx = fmaxf(s[0][r], s[1][r]);
#pragma unroll
            for (int msk = 8; msk >= 1; msk >>= 1)
                mx = fmaxf(mx, __shfl_xor(mx, msk, 32));
            const float mnew = fmaxf(m_run[r], mx);
            alpha[r] = __expf(m_run[r] - mnew);
            m_run[r] = mnew;
            const float e0 = __expf(s[0][r] - mnew);
            const float e1 = __expf(s[1][r] - mnew);
            s[0][r] = e0; s[1][r] = e1;
            float rs = e0 + e1;
#pragma unroll
            for (int msk = 8; msk >= 1; msk >>= 1)
                rs += __shfl_xor(rs, msk, 32);
            l_run[r] = l_run[r] * alpha[r] + rs;
        }

        // ---- P: C-layout -> LDS -> A-fragment layout (bf16)
#pragma unroll
        for (int j = 0; j < 2; ++j)
#pragma unroll
            for (int r = 0; r < 8; ++r)
                pbuf[wave][r + 8 * hi][j * 16 + ln] = f2bf(s[j][r]);
        asm volatile("s_wait_dscnt 0" ::: "memory");
        const bf16_t* pp = &pbuf[wave][ln][hi * 8];
        v16bf pf = join8(*(const v8bf*)pp, *(const v8bf*)(pp + 16));

        // ---- O = O*alpha + P(16x32) @ V(32x64)
#pragma unroll
        for (int j = 0; j < 4; ++j) {
#pragma unroll
            for (int r = 0; r < 8; ++r) o[j][r] *= alpha[r];
            o[j] = wmma_bf16(pf, vf[j], o[j]);
        }
    };

#pragma unroll 1
    for (int s0 = 0; s0 < T_DIM; s0 += 64) {
        chunk(s0, 0, 1);       // compute buf0, prefetch into buf1
        chunk(s0 + 32, 1, 0);  // compute buf1, prefetch into buf0
    }

    // ---- finalize: O /= l ; store bf16 to (t*B+b, h*D+d)
#pragma unroll
    for (int j = 0; j < 4; ++j)
#pragma unroll
        for (int r = 0; r < 8; ++r) {
            const float v = o[j][r] / l_run[r];
            const int t = t0 + r + 8 * hi;
            const int col = h * D_DIM + j * 16 + ln;
            attn_out[((size_t)t * B_DIM + b) * E_DIM + col] = f2bf(v);
        }
}

// ---------------------------------------------------------------- launcher
extern "C" void kernel_launch(void* const* d_in, const int* in_sizes, int n_in,
                              void* d_out, int out_size, void* d_ws,
                              size_t ws_size, hipStream_t stream) {
    (void)in_sizes; (void)n_in; (void)out_size; (void)ws_size;
    const float* query = (const float*)d_in[0];
    const float* abias = (const float*)d_in[1];
    const float* Wq = (const float*)d_in[2];
    const float* bq = (const float*)d_in[3];
    const float* Wk = (const float*)d_in[4];
    const float* bk = (const float*)d_in[5];
    const float* Wv = (const float*)d_in[6];
    const float* bv = (const float*)d_in[7];
    const float* Wo = (const float*)d_in[8];
    const float* bo = (const float*)d_in[9];
    float* out = (float*)d_out;

    const size_t NX = (size_t)T_DIM * B_DIM * E_DIM;  // 4,194,304
    const size_t NW = (size_t)E_DIM * E_DIM;          //   262,144
    const size_t NH = (size_t)B_DIM * H_DIM * T_DIM * D_DIM;  // 4,194,304

    char* ws = (char*)d_ws;
    size_t off = 0;
    bf16_t* Xbf = (bf16_t*)(ws + off); off += NX * 2;
    bf16_t* Wqb = (bf16_t*)(ws + off); off += NW * 2;
    bf16_t* Wkb = (bf16_t*)(ws + off); off += NW * 2;
    bf16_t* Wvb = (bf16_t*)(ws + off); off += NW * 2;
    bf16_t* Wob = (bf16_t*)(ws + off); off += NW * 2;
    bf16_t* Qb  = (bf16_t*)(ws + off); off += NH * 2;
    bf16_t* Kb  = (bf16_t*)(ws + off); off += NH * 2;
    bf16_t* VTb = (bf16_t*)(ws + off); off += NH * 2;
    bf16_t* Ab  = (bf16_t*)(ws + off); off += NX * 2;  // ~42 MB total

    // f32 -> bf16 conversions
    cvt_f32_bf16<<<(int)((NX + 255) / 256), 256, 0, stream>>>(query, Xbf, (int)NX);
    cvt_f32_bf16<<<(int)((NW + 255) / 256), 256, 0, stream>>>(Wq, Wqb, (int)NW);
    cvt_f32_bf16<<<(int)((NW + 255) / 256), 256, 0, stream>>>(Wk, Wkb, (int)NW);
    cvt_f32_bf16<<<(int)((NW + 255) / 256), 256, 0, stream>>>(Wv, Wvb, (int)NW);
    cvt_f32_bf16<<<(int)((NW + 255) / 256), 256, 0, stream>>>(Wo, Wob, (int)NW);

    const dim3 ggrid(64, 8);           // 8192/128 rows x 512/64 cols
    const float scale = 0.125f;        // D^-0.5
    gemm_wmma<<<ggrid, 256, 0, stream>>>(Xbf, Wqb, bq, scale, Qb, nullptr, 0);
    gemm_wmma<<<ggrid, 256, 0, stream>>>(Xbf, Wkb, bk, 1.0f, Kb, nullptr, 0);
    gemm_wmma<<<ggrid, 256, 0, stream>>>(Xbf, Wvb, bv, 1.0f, VTb, nullptr, 1);

    attn_wmma<<<dim3(T_DIM / 64, B_DIM * H_DIM), 128, 0, stream>>>(
        Qb, Kb, VTb, abias, Ab);

    gemm_wmma<<<ggrid, 256, 0, stream>>>(Ab, Wob, bo, 1.0f, nullptr, out, 2);
}
